// SelfAttention_77197742178633
// MI455X (gfx1250) — matrix operand order repaired and checked by reading, not compiled, not run on hardware
//
#include <hip/hip_runtime.h>
#include <hip/hip_bf16.h>
#include <math.h>

// MI455X (gfx1250) fused banded self-attention.
// Strategy:
//  - key = x@K done as bf16 WMMA with hi/lo error compensation (xh*Kh + xh*Kl + xl*Kh)
//    -> near-fp32 accuracy, 16x16x32 v_wmma throughput.
//  - Off-band logits underflow to exactly 0 after softmax (exp(-10000) == 0 in fp32),
//    so softmax runs over the <=12-wide band only, in exact fp32 VALU.
//  - Fused: key tile (32 rows x 512) lives in WMMA accumulators -> LDS -> band phase.
//    No HBM traffic for key at all.

typedef __attribute__((ext_vector_type(16))) __bf16 v16bf;
typedef __attribute__((ext_vector_type(8)))  __bf16 v8bf;
typedef __attribute__((ext_vector_type(8)))  float  v8f;

#define D_DIM 512
#define S_DIM 2048
#define BAND 12
#define HALF_BAND 6

__device__ __forceinline__ unsigned short f32_to_bf16(float f) {
  union { float f; unsigned u; } v; v.f = f;
  unsigned u = v.u;
  unsigned r = (u + 0x7FFFu + ((u >> 16) & 1u)) >> 16;  // round-to-nearest-even
  return (unsigned short)r;
}
__device__ __forceinline__ float bf16_to_f32(unsigned short h) {
  union { unsigned u; float f; } v; v.u = ((unsigned)h) << 16;
  return v.f;
}

// Split x (fp32) into bf16 hi + bf16 lo residual.
__global__ void convert_x_kernel(const float* __restrict__ x,
                                 unsigned short* __restrict__ xh,
                                 unsigned short* __restrict__ xl, int n) {
  int i = (blockIdx.x * blockDim.x + threadIdx.x) * 4;
  if (i + 3 < n) {
    float4 f = *(const float4*)(x + i);
    float fs[4] = {f.x, f.y, f.z, f.w};
#pragma unroll
    for (int k = 0; k < 4; ++k) {
      unsigned short h = f32_to_bf16(fs[k]);
      xh[i + k] = h;
      xl[i + k] = f32_to_bf16(fs[k] - bf16_to_f32(h));
    }
  }
}

// K[k][n] -> KT[n][k] in bf16 hi/lo, so WMMA B-fragments are contiguous loads.
__global__ void convert_kt_kernel(const float* __restrict__ K,
                                  unsigned short* __restrict__ kth,
                                  unsigned short* __restrict__ ktl) {
  int k = blockIdx.x * 16 + (threadIdx.x & 15);
  int n = blockIdx.y * 16 + (threadIdx.x >> 4);
  float f = K[k * D_DIM + n];
  unsigned short h = f32_to_bf16(f);
  kth[n * D_DIM + k] = h;
  ktl[n * D_DIM + k] = f32_to_bf16(f - bf16_to_f32(h));
}

// One workgroup (256 threads = 8 wave32) handles 32 flat rows (one batch, since
// S=2048 is a multiple of 32). Phase 1: key tile via WMMA. Phase 2: band softmax.
__global__ void __launch_bounds__(256)
fused_attn_kernel(const float* __restrict__ x,
                  const unsigned short* __restrict__ xh,
                  const unsigned short* __restrict__ xl,
                  const unsigned short* __restrict__ kth,
                  const unsigned short* __restrict__ ktl,
                  const float* __restrict__ temperature,
                  const float* __restrict__ bias,
                  float* __restrict__ out) {
  __shared__ float lds_key[32 * D_DIM];  // 64 KB of the WGP's 320 KB LDS

  const int lane = threadIdx.x & 31;
  const int wave = threadIdx.x >> 5;
  const int row0 = blockIdx.x * 32;   // flat row (b*S + s) of tile origin
  const int mt = wave >> 2;           // wave's 16-row M-tile (0..1)
  const int ng = wave & 3;            // wave's N-group: columns [ng*128, ng*128+128)
  const int lrow = lane & 15;
  const int lhalf = lane >> 4;

  // ---- Phase 1: key[row0:row0+32, :] = x @ K via bf16x2 compensated WMMA ----
  v8f acc[8] = {};
  const size_t arow = (size_t)(row0 + mt * 16 + lrow) * D_DIM;

#pragma unroll 1
  for (int ks = 0; ks < D_DIM / 32; ++ks) {
    // A fragment (16x32 bf16 layout): lanes 0-15 hold K 0-15, lanes 16-31 hold 16-31
    const int ka = ks * 32 + lhalf * 16;
    union { v16bf v; v8bf h[2]; } a_hi, a_lo;
    a_hi.h[0] = *(const v8bf*)(xh + arow + ka);
    a_hi.h[1] = *(const v8bf*)(xh + arow + ka + 8);
    a_lo.h[0] = *(const v8bf*)(xl + arow + ka);
    a_lo.h[1] = *(const v8bf*)(xl + arow + ka + 8);
    if (ks + 1 < D_DIM / 32) {  // gfx1250 global_prefetch of next A chunk
      __builtin_prefetch(xh + arow + ka + 32, 0, 3);
      __builtin_prefetch(xl + arow + ka + 32, 0, 3);
    }
    // B fragments (32x16 bf16): lane = column of KT, 16 contiguous K per lane-half
    const int kb = ks * 32 + lhalf * 16;
#pragma unroll
    for (int t = 0; t < 8; ++t) {
      const size_t brow = (size_t)((ng * 8 + t) * 16 + lrow) * D_DIM + kb;
      v16bf b_hi = *(const v16bf*)(kth + brow);
      v16bf b_lo = *(const v16bf*)(ktl + brow);
      acc[t] = __builtin_amdgcn_wmma_f32_16x16x32_bf16(false, a_hi.v, false, b_hi,
                                                       (short)0, acc[t], false, false);
      acc[t] = __builtin_amdgcn_wmma_f32_16x16x32_bf16(false, a_hi.v, false, b_lo,
                                                       (short)0, acc[t], false, false);
      acc[t] = __builtin_amdgcn_wmma_f32_16x16x32_bf16(false, a_lo.v, false, b_hi,
                                                       (short)0, acc[t], false, false);
    }
  }

  // Spill key tile to LDS (C/D layout: lane l holds col l%16, rows (l/16)*8 + r)
#pragma unroll
  for (int t = 0; t < 8; ++t) {
    const int ncol = (ng * 8 + t) * 16 + lrow;
#pragma unroll
    for (int r = 0; r < 8; ++r) {
      lds_key[(mt * 16 + lhalf * 8 + r) * D_DIM + ncol] = acc[t][r];
    }
  }
  __syncthreads();

  // ---- Phase 2: banded logits + softmax + weighted sum (exact fp32) ----
  const float temp = temperature[0];
  const float bs = bias[0];
  const int b = row0 >> 11;        // row0 / S_DIM
  const int sbase = row0 & (S_DIM - 1);
  const float* __restrict__ xb = x + (size_t)b * S_DIM * D_DIM;

  for (int q = 0; q < 4; ++q) {
    const int i = wave * 4 + q;    // row within tile, one row per wave iteration
    const int s = sbase + i;

    float kreg[16];
#pragma unroll
    for (int c = 0; c < 16; ++c) kreg[c] = lds_key[i * D_DIM + lane + 32 * c];

    float dots[BAND];
#pragma unroll 1
    for (int j = 0; j < BAND; ++j) {
      const int t = s - HALF_BAND + j;
      float p = 0.0f;
      if (t >= 0 && t < S_DIM) {   // uniform across the wave
        const float* xr = xb + (size_t)t * D_DIM;
#pragma unroll
        for (int c = 0; c < 16; ++c) p += kreg[c] * xr[lane + 32 * c];
      }
#pragma unroll
      for (int off = 16; off > 0; off >>= 1) p += __shfl_xor(p, off, 32);
      dots[j] = p;                 // full dot broadcast to all 32 lanes
    }

    float lg[BAND];
    float mx = -1e30f;
#pragma unroll
    for (int j = 0; j < BAND; ++j) {
      const int t = s - HALF_BAND + j;
      if (t >= 0 && t < S_DIM) {
        lg[j] = tanhf(dots[j] + bs) * temp;
        mx = fmaxf(mx, lg[j]);
      } else {
        lg[j] = -1e30f;
      }
    }
    float wsum = 0.0f, wgt[BAND];
#pragma unroll
    for (int j = 0; j < BAND; ++j) {
      const int t = s - HALF_BAND + j;
      const float e = (t >= 0 && t < S_DIM) ? __expf(lg[j] - mx) : 0.0f;
      wgt[j] = e;
      wsum += e;
    }
    const float inv = 1.0f / wsum;

    float o[16];
#pragma unroll
    for (int c = 0; c < 16; ++c) o[c] = 0.0f;
#pragma unroll 1
    for (int j = 0; j < BAND; ++j) {
      const int t = s - HALF_BAND + j;
      if (t >= 0 && t < S_DIM) {
        const float* xr = xb + (size_t)t * D_DIM;
        const float w = wgt[j];
#pragma unroll
        for (int c = 0; c < 16; ++c) o[c] += w * xr[lane + 32 * c];
      }
    }
    float* orow = out + ((size_t)b * S_DIM + s) * D_DIM;
#pragma unroll
    for (int c = 0; c < 16; ++c) orow[lane + 32 * c] = o[c] * inv;
  }
}

extern "C" void kernel_launch(void* const* d_in, const int* in_sizes, int n_in,
                              void* d_out, int out_size, void* d_ws, size_t ws_size,
                              hipStream_t stream) {
  const float* x = (const float*)d_in[0];
  const float* K = (const float*)d_in[1];
  const float* temperature = (const float*)d_in[2];
  const float* bias = (const float*)d_in[3];
  float* out = (float*)d_out;

  const int R = in_sizes[0] / D_DIM;  // B*S flat rows (16384)

  // Workspace layout: xh, xl [R*D] bf16 ; KTh, KTl [D*D] bf16  (~34.6 MB total)
  unsigned short* xh = (unsigned short*)d_ws;
  unsigned short* xl = xh + (size_t)R * D_DIM;
  unsigned short* kth = xl + (size_t)R * D_DIM;
  unsigned short* ktl = kth + (size_t)D_DIM * D_DIM;

  const int nx = R * D_DIM;
  convert_x_kernel<<<(nx / 4 + 255) / 256, 256, 0, stream>>>(x, xh, xl, nx);

  dim3 gk(D_DIM / 16, D_DIM / 16);
  convert_kt_kernel<<<gk, 256, 0, stream>>>(K, kth, ktl);

  fused_attn_kernel<<<R / 32, 256, 0, stream>>>(x, xh, xl, kth, ktl,
                                                temperature, bias, out);
}
